// Head_996432413512
// MI455X (gfx1250) — matrix operand order repaired and checked by reading, not compiled, not run on hardware
//
#include <hip/hip_runtime.h>
#include <hip/hip_bf16.h>

// ---------------------------------------------------------------------------
// RWKV-style WKV + LIF spiking head for MI455X (gfx1250, wave32, WMMA).
//   B=16, T=2048, D=1024.
//   Stage 1: token-shift + pack x to bf16.
//   Stage 2: transpose-convert the 4 weights into one bf16 [1024][4096] matrix.
//   Stage 3: single fused GEMM (32768x4096x1024) with per-weight activation
//            epilogue via v_wmma_f32_16x16x32_bf16, double-buffered LDS fed by
//            GLOBAL_LOAD_ASYNC_TO_LDS_B128 (ASYNCcnt) when available.
//   Stage 4: sequential WKV+LIF scan over T per (b,d) channel (NT streaming).
// ---------------------------------------------------------------------------

typedef __attribute__((ext_vector_type(16))) __bf16 v16bf;
typedef __attribute__((ext_vector_type(8)))  float  v8f;
typedef __attribute__((ext_vector_type(4)))  int    v4i_;

#define SB 16
#define ST 2048
#define SD 1024
#define SM (SB * ST)          // 32768 GEMM rows
#define SN 4096               // 4 fused weight outputs
#define KTILE 64
#define NKT (SD / KTILE)      // 16 K-iterations

#if __has_builtin(__builtin_amdgcn_global_load_async_to_lds_b128)
#define HAVE_ASYNC_LDS 1
#else
#define HAVE_ASYNC_LDS 0
#endif

// 16-byte global -> LDS copy. Async (VGPR-bypassing, ASYNCcnt-tracked) when
// the toolchain exposes the gfx1250 builtin; plain b128 staging otherwise.
// Builtin signature (from clang diagnostic): param0 = v4i AS(1)* (global src),
// param1 = v4i AS(3)* (LDS dst), then imm offset + imm cpol.
__device__ __forceinline__ void cp_g2l_16(void* lds, const void* g) {
#if HAVE_ASYNC_LDS
  __builtin_amdgcn_global_load_async_to_lds_b128(
      (__attribute__((address_space(1))) v4i_*)(g),
      (__attribute__((address_space(3))) v4i_*)(lds),
      /*imm offset=*/0, /*cpol=*/0);
#else
  *(uint4*)lds = *(const uint4*)g;
#endif
}

// Drain outstanding async LDS fills; must precede the barrier signal so no
// consumer reads a tile before the async writes land.
__device__ __forceinline__ void wait_g2l() {
#if HAVE_ASYNC_LDS
#if __has_builtin(__builtin_amdgcn_s_wait_asynccnt)
  __builtin_amdgcn_s_wait_asynccnt(0);
#else
  asm volatile("s_wait_asynccnt 0x0" ::: "memory");
#endif
#endif
}

// ---------------- Stage 1: token shift + bf16 pack -------------------------
__global__ __launch_bounds__(256) void k_tokenshift(const float* __restrict__ emb,
                                                    const float* __restrict__ mix,
                                                    __bf16* __restrict__ xbf) {
  size_t i = (size_t)blockIdx.x * 256 + threadIdx.x;     // < SM*SD
  int d = (int)(i & (SD - 1));
  size_t t = (i >> 10) & (ST - 1);
  float e  = emb[i];
  float sh = (t == 0) ? 0.0f : emb[i - SD];
  float mx = mix[d];
  xbf[i] = (__bf16)(mx * e + (1.0f - mx) * sh);
}

// ---------------- Stage 2: weight transpose + bf16 pack --------------------
// wt[d][w*1024+e] = (bf16) W_w[e][d]  -> one [1024][4096] B matrix,
// write-side coalesced; read lines stay resident in the 192 MB L2.
__global__ __launch_bounds__(256) void k_wconv(const float* __restrict__ W0,
                                               const float* __restrict__ W1,
                                               const float* __restrict__ W2,
                                               const float* __restrict__ W3,
                                               __bf16* __restrict__ wt) {
  size_t i = (size_t)blockIdx.x * 256 + threadIdx.x;     // < 4*SD*SD
  int w = (int)(i >> 20);
  size_t r = i & ((1u << 20) - 1);
  int d = (int)(r >> 10);
  int e = (int)(r & (SD - 1));
  const float* Wp = (w == 0) ? W0 : (w == 1) ? W1 : (w == 2) ? W2 : W3;
  wt[((size_t)d << 12) + ((size_t)w << 10) + e] = (__bf16)Wp[((size_t)e << 10) + d];
}

// ---------------- Stage 3: fused GEMM + activation (WMMA bf16) -------------
// out[m,n] = act_w( sum_k X[m,k] * Wt[k,n] ),  M=32768, N=4096, K=1024.
// Block tile 128(M) x 128(N), K-step 64, double-buffered LDS.
// 8 waves: 4 along M x 2 along N; wave tile 32x64 = 2x4 WMMA accumulators.
// mode = n>>10: 0 sigmoid (r), 1 exp(min(.,10)) (ek), 2 identity (v), 3 exp (td)
__global__ __launch_bounds__(256) void k_gemm4_act(const __bf16* __restrict__ X,
                                                   const __bf16* __restrict__ Wt,
                                                   float* __restrict__ go) {
  __shared__ __bf16 As[2][128][KTILE + 8];   // 2 x 18 KB
  __shared__ __bf16 Bs[2][KTILE][128 + 8];   // 2 x 17 KB

  const int tid  = threadIdx.x;
  const int lane = tid & 31;
  const int wave = tid >> 5;
  const int wm   = wave & 3;        // M offset wm*32
  const int wn   = wave >> 2;       // N offset wn*64

  const size_t mBase = (size_t)blockIdx.y * 128;
  const int    nBase = blockIdx.x * 128;            // 0..4095, one weight per tile
  const int    mode  = nBase >> 10;                 // uniform per block
  const int    eBase = nBase & (SD - 1);
  float* out = go + (size_t)mode * ((size_t)SM * SD);

  v8f acc[2][4] = {};

  // Tile loaders: 1024 b128 chunks per tile, 4 per thread, 16B aligned.
  auto loadTiles = [&](int kk, int buf) {
    #pragma unroll
    for (int q = 0; q < 4; ++q) {
      const int c  = tid + 256 * q;
      const int ar = c >> 3, ac = (c & 7) * 8;        // A: 128 rows x 64 K
      cp_g2l_16(&As[buf][ar][ac], X + (mBase + ar) * SD + kk + ac);
      const int br = c >> 4, bc = (c & 15) * 8;       // B: 64 K x 128 N
      cp_g2l_16(&Bs[buf][br][bc], Wt + (size_t)(kk + br) * SN + nBase + bc);
    }
  };

  loadTiles(0, 0);
  wait_g2l();
  __syncthreads();

  const int r15   = lane & 15;
  const int khalf = (lane >> 4) * 8;   // ISA A layout: lane halves split K

  for (int kc = 0; kc < NKT; ++kc) {
    const int buf = kc & 1;
    if (kc + 1 < NKT) loadTiles((kc + 1) * KTILE, buf ^ 1);
    if (kc + 2 < NKT) {                // warm GL2 two tiles ahead
      __builtin_prefetch(X + (mBase + (tid >> 3)) * SD + (kc + 2) * KTILE + (tid & 7) * 8, 0, 1);
      __builtin_prefetch(Wt + (size_t)((kc + 2) * KTILE + (tid >> 4)) * SN + nBase + (tid & 15) * 8, 0, 1);
    }

    #pragma unroll
    for (int kh = 0; kh < 2; ++kh) {   // two K=32 sub-chunks
      const int k0 = kh * 32;
      // A fragments: lane L<16 -> row M=L, K {0..7}u{16..23};
      //              lane L>=16 -> row M=L-16, K {8..15}u{24..31}.
      v16bf afr[2];
      #pragma unroll
      for (int i = 0; i < 2; ++i) {
        const int r = wm * 32 + i * 16 + r15;
        #pragma unroll
        for (int q = 0; q < 8; ++q) {
          afr[i][q]     = As[buf][r][k0 + khalf + q];
          afr[i][8 + q] = As[buf][r][k0 + 16 + khalf + q];
        }
      }
      // B fragments: column N = lane&15, K in same order as A.
      v16bf bfr[4];
      #pragma unroll
      for (int j = 0; j < 4; ++j) {
        const int n0 = wn * 64 + j * 16;
        #pragma unroll
        for (int q = 0; q < 8; ++q) {
          bfr[j][q]     = Bs[buf][k0 + khalf + q][n0 + r15];
          bfr[j][8 + q] = Bs[buf][k0 + 16 + khalf + q][n0 + r15];
        }
      }
      #pragma unroll
      for (int i = 0; i < 2; ++i)
        #pragma unroll
        for (int j = 0; j < 4; ++j)
          acc[i][j] = __builtin_amdgcn_wmma_f32_16x16x32_bf16(
              false, afr[i], false, bfr[j], (short)0, acc[i][j], false, false);
    }
    wait_g2l();          // async fills of buf^1 must land before the barrier
    __syncthreads();
  }

  // Epilogue. C/D layout: VGPR q -> M=q (lanes 0-15) / q+8 (lanes 16-31),
  // N = lane&15. NT stores: planes are streaming, keep x resident in L2.
  const int mhalf = (lane >> 4) * 8;
  #pragma unroll
  for (int i = 0; i < 2; ++i) {
    #pragma unroll
    for (int j = 0; j < 4; ++j) {
      const size_t row0 = mBase + (size_t)(wm * 32 + i * 16 + mhalf);
      const int    ecol = eBase + wn * 64 + j * 16 + r15;
      #pragma unroll
      for (int q = 0; q < 8; ++q) {
        float c = acc[i][j][q];
        float val;
        if (mode == 0)      val = 1.0f / (1.0f + __expf(-c));        // r
        else if (mode == 1) val = __expf(fminf(c, 10.0f));           // exp_k
        else if (mode == 2) val = c;                                 // v
        else                val = __expf(c);                         // td
        __builtin_nontemporal_store(val, &out[(row0 + q) * SD + ecol]);
      }
    }
  }
}

// ---------------- Stage 4: WKV + LIF scan over T ---------------------------
__global__ __launch_bounds__(256) void k_scan(const float* __restrict__ r_,
                                              const float* __restrict__ ek_,
                                              const float* __restrict__ v_,
                                              const float* __restrict__ td_,
                                              const float* __restrict__ gdecay,
                                              float* __restrict__ out) {
  const int g = blockIdx.x * 256 + threadIdx.x;  // 0 .. B*D-1
  const int b = g >> 10;
  const int d = g & (SD - 1);
  const float gd = __expf(gdecay[d]);

  float a = 0.0f, bb = 0.0f, mem = 0.0f;
  size_t idx = ((size_t)b * ST) * SD + d;
  for (int t = 0; t < ST; ++t, idx += SD) {
    const float ek = __builtin_nontemporal_load(&ek_[idx]);
    const float vv = __builtin_nontemporal_load(&v_[idx]);
    const float td = __builtin_nontemporal_load(&td_[idx]);
    const float rr = __builtin_nontemporal_load(&r_[idx]);
    a  = ek * vv + td * a;
    bb = ek + td * bb;
    const float wkv  = (gd * ek * vv + a) / (gd * ek + bb + 1e-6f);
    const float x_in = rr * wkv;
    const float reset = (mem > 0.2f) ? 1.0f : 0.0f;   // from previous mem
    mem = 0.3f * mem + x_in - reset * 0.2f;
    const float spk = ((mem - 0.2f) > 0.0f) ? 1.0f : 0.0f;
    __builtin_nontemporal_store(spk, &out[idx]);
  }
}

// ---------------------------------------------------------------------------
extern "C" void kernel_launch(void* const* d_in, const int* in_sizes, int n_in,
                              void* d_out, int out_size, void* d_ws, size_t ws_size,
                              hipStream_t stream) {
  const float* emb  = (const float*)d_in[0];
  const float* W_R  = (const float*)d_in[1];
  const float* W_K  = (const float*)d_in[2];
  const float* W_V  = (const float*)d_in[3];
  const float* W_td = (const float*)d_in[4];
  const float* gdec = (const float*)d_in[5];
  const float* mix  = (const float*)d_in[6];
  float* out = (float*)d_out;

  char* ws = (char*)d_ws;
  __bf16* xbf = (__bf16*)ws;                                   // 64 MB
  __bf16* wt  = (__bf16*)(ws + (size_t)64 * 1024 * 1024);      //  8 MB
  float*  go  = (float*)(ws + (size_t)72 * 1024 * 1024);       // 4 x 128 MB
  const size_t plane = (size_t)SM * SD;                        // 2^25 elements

  // Stage 1: token shift + pack. SM*SD / 256 = 131072 blocks.
  k_tokenshift<<<131072, 256, 0, stream>>>(emb, mix, xbf);

  // Stage 2: weight transpose + pack. 4*2^20 / 256 = 16384 blocks.
  k_wconv<<<16384, 256, 0, stream>>>(W_R, W_K, W_V, W_td, wt);

  // Stage 3: single fused GEMM. Grid: 32 N-tiles x 256 M-tiles.
  dim3 gg(SN / 128, SM / 128);
  k_gemm4_act<<<gg, 256, 0, stream>>>(xbf, wt, go);

  // Stage 4: scan. B*D = 16384 channels -> 64 blocks.
  k_scan<<<64, 256, 0, stream>>>(go + 0 * plane,   // r
                                 go + 1 * plane,   // exp_k
                                 go + 2 * plane,   // v
                                 go + 3 * plane,   // td
                                 gdec, out);
}
